// GenericTFB_61538291417596
// MI455X (gfx1250) — matrix operand order repaired and checked by reading, not compiled, not run on hardware
//
#include <hip/hip_runtime.h>
#include <cmath>

// ---------------- types ----------------
typedef __bf16 bf16;
typedef __attribute__((ext_vector_type(16))) __bf16 v16bf;
typedef __attribute__((ext_vector_type(8)))  __bf16 v8bf;
typedef __attribute__((ext_vector_type(8)))  float  v8f;
typedef unsigned int v4u_t __attribute__((ext_vector_type(4)));
typedef int          v8i_t __attribute__((ext_vector_type(8)));
typedef int          v4i_t __attribute__((ext_vector_type(4)));

// ---------------- problem constants ----------------
static constexpr int Bb  = 3;
static constexpr int S   = 64;
static constexpr int E   = 2048;
static constexpr int H   = 64;
static constexpr int Dh  = 32;
static constexpr int HID = 8192;
static constexpr int M   = Bb * S * S;     // 12288 rows
static constexpr int NQKV = 3 * H * Dh;    // 6144

// ---------------- CDNA5 data-movement feature detection ----------------
#if defined(__has_builtin)
#  if __has_builtin(__builtin_amdgcn_tensor_load_to_lds) && __has_builtin(__builtin_amdgcn_s_wait_tensorcnt)
#    define GEMM_USE_TDM 1
#  elif __has_builtin(__builtin_amdgcn_global_load_async_to_lds_b128)
#    define GEMM_USE_ASYNC 1
#  endif
#endif

#if defined(GEMM_USE_TDM)
// ROCm 7.2 clang-22 declares the 5-arg builtin; the therock-10.0 toolchain
// (which ships amd_gfx1250_TDM.h) declares a 6-arg form.
#  if __has_include(<hip/amd_detail/amd_gfx1250_TDM.h>)
#    define TDM_LOAD(g0, g1, z4) \
        __builtin_amdgcn_tensor_load_to_lds((g0), (g1), (z4), (z4), (v8i_t){0,0,0,0,0,0,0,0}, 0)
#  else
#    define TDM_LOAD(g0, g1, z4) \
        __builtin_amdgcn_tensor_load_to_lds((g0), (g1), (z4), (z4), 0)
#  endif
#endif

__device__ __forceinline__ unsigned lds_addr_of(const void* p) {
    // generic -> LDS address space, then to the 32-bit LDS byte offset
    return (unsigned)(unsigned long long)(const __attribute__((address_space(3))) char*)p;
}

// ---------------- WMMA helpers ----------------
__device__ __forceinline__ v8f wmma_bf16(v16bf a, v16bf b, v8f c) {
    return __builtin_amdgcn_wmma_f32_16x16x32_bf16(false, a, false, b, (short)0, c, false, false);
}

// A fragment: 16x32 bf16 tile, row-major, leading dim ld (elements).
__device__ __forceinline__ v16bf load_frag_a(const bf16* base, int ld) {
    const int lane = threadIdx.x & 31;
    const int row  = lane & 15;
    const int koff = (lane >> 4) << 3;
    const bf16* p = base + (size_t)row * ld + koff;
    union { v16bf v; v8bf h[2]; } u;
    u.h[0] = *(const v8bf*)(p);
    u.h[1] = *(const v8bf*)(p + 16);
    return u.v;
}

// B fragment: 32x16 (KxN); source stored N-major (row n holds contiguous K).
__device__ __forceinline__ v16bf load_frag_b(const bf16* base, int ld) {
    const int lane  = threadIdx.x & 31;
    const int col   = lane & 15;
    const int kbase = (lane >> 4) << 3;
    const bf16* p = base + (size_t)col * ld + kbase;
    union { v16bf v; v8bf h[2]; } u;
    u.h[0] = *(const v8bf*)(p);
    u.h[1] = *(const v8bf*)(p + 16);
    return u.v;
}

// ---------------- weight conversion kernels ----------------
__global__ __launch_bounds__(256) void cvt_bf16_kernel(const float* __restrict__ in,
                                                       bf16* __restrict__ out, size_t n) {
    size_t i = (size_t)blockIdx.x * 256 + threadIdx.x;
    size_t stride = (size_t)gridDim.x * 256;
    for (; i < n; i += stride) out[i] = (bf16)in[i];
}

// in: R x C f32 row-major  ->  out: C x R bf16 row-major
__global__ __launch_bounds__(256) void transpose_cvt_kernel(const float* __restrict__ in,
                                                            bf16* __restrict__ out,
                                                            int R, int C) {
    __shared__ float tile[32][33];
    const int bx = blockIdx.x * 32;
    const int by = blockIdx.y * 32;
    const int tx = threadIdx.x & 31;
    const int ty = threadIdx.x >> 5;
    #pragma unroll
    for (int yy = ty; yy < 32; yy += 8)
        tile[yy][tx] = in[(size_t)(by + yy) * C + (bx + tx)];
    __syncthreads();
    #pragma unroll
    for (int yy = ty; yy < 32; yy += 8)
        out[(size_t)(bx + yy) * R + (by + tx)] = (bf16)tile[tx][yy];
}

// ---------------- LayerNorm (row = 2048 f32) ----------------
__global__ __launch_bounds__(256) void ln_kernel(const float* __restrict__ in,
                                                 bf16* __restrict__ out_bf,
                                                 float* __restrict__ out_f,
                                                 int ncols) {
    const int row = blockIdx.x;
    const int tid = threadIdx.x;
    const float* x = in + (size_t)row * ncols;
    float v[8];
    float s = 0.f, ss = 0.f;
    #pragma unroll
    for (int i = 0; i < 8; ++i) {
        float t = x[tid + i * 256];
        v[i] = t; s += t; ss += t * t;
    }
    __shared__ float r1[256], r2[256];
    r1[tid] = s; r2[tid] = ss;
    __syncthreads();
    for (int off = 128; off > 0; off >>= 1) {
        if (tid < off) { r1[tid] += r1[tid + off]; r2[tid] += r2[tid + off]; }
        __syncthreads();
    }
    const float mean = r1[0] / (float)ncols;
    const float var  = r2[0] / (float)ncols - mean * mean;
    const float rstd = rsqrtf(var + 1e-5f);
    #pragma unroll
    for (int i = 0; i < 8; ++i) {
        const int c = tid + i * 256;
        const float z = (v[i] - mean) * rstd;
        out_bf[(size_t)row * ncols + c] = (bf16)z;
        if (out_f) out_f[(size_t)row * ncols + c] = z;
    }
}

// ---------------- tiled bf16 WMMA GEMM (double-buffered TDM/async staging) ----
// C[M,N] = A[M,K] (bf16 row-major) x W[N,K]^T (bf16, N-major) + epilogue
// mode 0: bf16 out          mode 1: f32 out = acc + resid
// mode 2: bf16 gelu(acc+b)  mode 3: f32 out = acc + bias + resid
#define BM 128
#define BN 128
#define BK 32
#define LDT 48   // padded LDS leading dim: 96B rows, 16B aligned

__global__ __launch_bounds__(256) void gemm_bf16_kernel(
    const bf16* __restrict__ A, const bf16* __restrict__ W,
    int Mm, int Nn, int Kk,
    const float* __restrict__ bias, const float* __restrict__ resid,
    bf16* __restrict__ out_bf, float* __restrict__ out_f, int mode)
{
    __shared__ bf16 As[2][BM * LDT];
    __shared__ bf16 Bs[2][BN * LDT];

    const int tid  = threadIdx.x;
    const int lane = tid & 31;
    const int wave = tid >> 5;     // 0..7
    const int wm   = wave >> 2;    // 0..1 : 64-row group
    const int wn   = wave & 3;     // 0..3 : 32-col group
    const int rowBlk = blockIdx.y * BM;
    const int colBlk = blockIdx.x * BN;

    const v8f vzero = {};
    v8f acc[4][2];
    #pragma unroll
    for (int mt = 0; mt < 4; ++mt)
        #pragma unroll
        for (int nt = 0; nt < 2; ++nt) acc[mt][nt] = vzero;

    const bf16* Ab = A + (size_t)rowBlk * Kk;
    const bf16* Wb = W + (size_t)colBlk * Kk;

    auto compute = [&](int buf) {
        v16bf af[4], bfr[2];
        #pragma unroll
        for (int mt = 0; mt < 4; ++mt)
            af[mt] = load_frag_a(&As[buf][(wm * 64 + mt * 16) * LDT], LDT);
        #pragma unroll
        for (int nt = 0; nt < 2; ++nt)
            bfr[nt] = load_frag_b(&Bs[buf][(wn * 32 + nt * 16) * LDT], LDT);
        #pragma unroll
        for (int mt = 0; mt < 4; ++mt)
            #pragma unroll
            for (int nt = 0; nt < 2; ++nt)
                acc[mt][nt] = wmma_bf16(af[mt], bfr[nt], acc[mt][nt]);
    };

#if defined(GEMM_USE_TDM)
    // ---- Tensor Data Mover staging: wave 0 issues one 2D tile descriptor per
    // matrix per K-step; LDS row padding (32 elem data + 16 elem pad) done by TDM.
    const unsigned asBase = lds_addr_of(&As[0][0]);
    const unsigned bsBase = lds_addr_of(&Bs[0][0]);

    auto tdm_issue = [&](int buf, int k0) {
        const v4i_t z4 = {0, 0, 0, 0};
        // group1 word0: data_size=2B(1), pad_enable, pad_interval=16 DW(3), pad_amount=8 DW(7)
        const int w0 = (1 << 16) | (1 << 20) | (3 << 22) | (7 << 25);
        {   // A tile: tile 32x128 at (rowBlk, k0)
            unsigned long long ga = (unsigned long long)(size_t)(const void*)(Ab + k0);
            v4u_t g0; g0[0] = 1u;
            g0[1] = asBase + (unsigned)buf * (BM * LDT * 2);
            g0[2] = (unsigned)ga;
            g0[3] = (unsigned)((ga >> 32) & 0x01FFFFFFull) | (2u << 30);   // type=2
            const int td0 = Kk - k0, td1 = Mm - rowBlk;
            v8i_t g1;
            g1[0] = w0;
            g1[1] = (td0 & 0xffff) << 16;
            g1[2] = ((td0 >> 16) & 0xffff) | ((td1 & 0xffff) << 16);
            g1[3] = ((td1 >> 16) & 0xffff) | (BK << 16);                    // tile_dim0=32
            g1[4] = BM;                                                     // tile_dim1=128
            g1[5] = Kk;                                                     // dim0 stride
            g1[6] = 0; g1[7] = 0;
            TDM_LOAD(g0, g1, z4);
        }
        {   // B tile: tile 32x128 at (colBlk, k0)
            unsigned long long gb = (unsigned long long)(size_t)(const void*)(Wb + k0);
            v4u_t g0; g0[0] = 1u;
            g0[1] = bsBase + (unsigned)buf * (BN * LDT * 2);
            g0[2] = (unsigned)gb;
            g0[3] = (unsigned)((gb >> 32) & 0x01FFFFFFull) | (2u << 30);
            const int td0 = Kk - k0, td1 = Nn - colBlk;
            v8i_t g1;
            g1[0] = w0;
            g1[1] = (td0 & 0xffff) << 16;
            g1[2] = ((td0 >> 16) & 0xffff) | ((td1 & 0xffff) << 16);
            g1[3] = ((td1 >> 16) & 0xffff) | (BK << 16);
            g1[4] = BN;
            g1[5] = Kk;
            g1[6] = 0; g1[7] = 0;
            TDM_LOAD(g0, g1, z4);
        }
    };

    if (wave == 0) tdm_issue(0, 0);
    int buf = 0;
    for (int k0 = 0; k0 < Kk; k0 += BK, buf ^= 1) {
        if (wave == 0) {
            if (k0 + BK < Kk) {
                tdm_issue(buf ^ 1, k0 + BK);          // prefetch next tile
                __builtin_amdgcn_s_wait_tensorcnt(2); // current tile complete (in-order)
            } else {
                __builtin_amdgcn_s_wait_tensorcnt(0);
            }
        }
        __syncthreads();
        compute(buf);
        __syncthreads();
    }
#elif defined(GEMM_USE_ASYNC)
    // ---- async global->LDS staging (ASYNCcnt), double buffered
    auto stage_async = [&](int buf, int k0) {
        #pragma unroll
        for (int p = 0; p < 2; ++p) {
            const int idx = tid + p * 256;
            const int r   = idx >> 2;
            const int c   = (idx & 3) << 3;
            __builtin_amdgcn_global_load_async_to_lds_b128(
                (__attribute__((address_space(1))) void*)(void*)(Ab + (size_t)r * Kk + k0 + c),
                (__attribute__((address_space(3))) void*)&As[buf][r * LDT + c], 0, 0);
            __builtin_amdgcn_global_load_async_to_lds_b128(
                (__attribute__((address_space(1))) void*)(void*)(Wb + (size_t)r * Kk + k0 + c),
                (__attribute__((address_space(3))) void*)&Bs[buf][r * LDT + c], 0, 0);
        }
    };
    stage_async(0, 0);
    int buf = 0;
    for (int k0 = 0; k0 < Kk; k0 += BK, buf ^= 1) {
        if (k0 + BK < Kk) {
            stage_async(buf ^ 1, k0 + BK);
            asm volatile("s_wait_asynccnt 4" ::: "memory");  // 4 newest still in flight
        } else {
            asm volatile("s_wait_asynccnt 0" ::: "memory");
        }
        __syncthreads();
        compute(buf);
        __syncthreads();
    }
#else
    // ---- synchronous fallback
    for (int k0 = 0; k0 < Kk; k0 += BK) {
        __syncthreads();
        v8bf tA[2], tB[2];
        #pragma unroll
        for (int p = 0; p < 2; ++p) {
            const int idx = tid + p * 256;
            const int r   = idx >> 2;
            const int c   = (idx & 3) << 3;
            tA[p] = *(const v8bf*)(Ab + (size_t)r * Kk + k0 + c);
            tB[p] = *(const v8bf*)(Wb + (size_t)r * Kk + k0 + c);
        }
        #pragma unroll
        for (int p = 0; p < 2; ++p) {
            const int idx = tid + p * 256;
            const int r   = idx >> 2;
            const int c   = (idx & 3) << 3;
            *(v8bf*)&As[0][r * LDT + c] = tA[p];
            *(v8bf*)&Bs[0][r * LDT + c] = tB[p];
        }
        if (k0 + BK < Kk) {
            __builtin_prefetch(Ab + (size_t)(tid >> 1) * Kk + k0 + BK, 0, 1);
            __builtin_prefetch(Wb + (size_t)(tid >> 1) * Kk + k0 + BK, 0, 1);
        }
        __syncthreads();
        compute(0);
    }
#endif

    // ---- epilogue: lane holds col (lane&15), rows i + 8*(lane>>4)
    const int rl = (lane >> 4) << 3;
    const int cl = lane & 15;
    #pragma unroll
    for (int mt = 0; mt < 4; ++mt) {
        #pragma unroll
        for (int nt = 0; nt < 2; ++nt) {
            #pragma unroll
            for (int i = 0; i < 8; ++i) {
                const int r = rowBlk + wm * 64 + mt * 16 + rl + i;
                const int c = colBlk + wn * 32 + nt * 16 + cl;
                const size_t off = (size_t)r * Nn + c;
                float x = acc[mt][nt][i];
                if (mode == 0) {
                    out_bf[off] = (bf16)x;
                } else if (mode == 1) {
                    out_f[off] = x + resid[off];
                } else if (mode == 2) {
                    x += bias[c];
                    const float g = 0.5f * x * (1.0f + erff(x * 0.70710678118654752f));
                    out_bf[off] = (bf16)g;
                } else {
                    out_f[off] = x + bias[c] + resid[off];
                }
            }
        }
    }
}

// ---------------- fused attention ----------------
__global__ __launch_bounds__(128) void attn_kernel(const bf16* __restrict__ qkv,
                                                   bf16* __restrict__ ctx) {
    const int blk  = blockIdx.x;          // t*4096 + iPos*64 + h
    const int t    = blk >> 12;
    const int iPos = (blk >> 6) & 63;
    const int h    = blk & 63;
    const int tid  = threadIdx.x;
    const int lane = tid & 31;
    const int wave = tid >> 5;            // 0..3

    __shared__ float sc[4][16][68];
    __shared__ bf16  at[4][16][64];
    __shared__ bf16  vt[32][64];          // vt[d][j]

    const size_t colOff = (size_t)t * 2048 + (size_t)h * 32;
    const bf16* qb = qkv + (size_t)(0 * 4096 + iPos * 64) * NQKV + colOff;
    const bf16* kb = qkv + (size_t)(1 * 4096 + iPos * 64) * NQKV + colOff;
    const bf16* vb = qkv + (size_t)(2 * 4096 + iPos * 64) * NQKV + colOff;

    for (int e = tid; e < 64 * 32; e += 128) {
        const int j = e >> 5, d = e & 31;
        vt[d][j] = vb[(size_t)j * NQKV + d];
    }

    v16bf qf;
    {
        const int row  = wave * 16 + (lane & 15);
        const int koff = (lane >> 4) << 3;
        const bf16* p = qb + (size_t)row * NQKV + koff;
        union { v16bf v; v8bf h2[2]; } u;
        u.h2[0] = *(const v8bf*)(p);
        u.h2[1] = *(const v8bf*)(p + 16);
        qf = u.v;
    }

    const float scale = 0.022097086912079608f; // 1/sqrt(E)
    const v8f vzero = {};

    #pragma unroll
    for (int nt = 0; nt < 4; ++nt) {
        v16bf kf;
        {
            const int col   = nt * 16 + (lane & 15);
            const int kbase = (lane >> 4) << 3;
            const bf16* p = kb + (size_t)col * NQKV + kbase;
            union { v16bf v; v8bf h2[2]; } u;
            u.h2[0] = *(const v8bf*)(p);
            u.h2[1] = *(const v8bf*)(p + 16);
            kf = u.v;
        }
        v8f a = wmma_bf16(qf, kf, vzero);
        const int rl = (lane >> 4) << 3;
        const int cl = lane & 15;
        #pragma unroll
        for (int i = 0; i < 8; ++i)
            sc[wave][rl + i][nt * 16 + cl] = a[i] * scale;
    }
    __syncthreads();

    if (lane < 16) {
        float mx = -3.4e38f;
        for (int c = 0; c < 64; ++c) mx = fmaxf(mx, sc[wave][lane][c]);
        float s = 0.f;
        for (int c = 0; c < 64; ++c) s += __expf(sc[wave][lane][c] - mx);
        const float inv = 1.0f / s;
        for (int c = 0; c < 64; ++c)
            at[wave][lane][c] = (bf16)(__expf(sc[wave][lane][c] - mx) * inv);
    }
    __syncthreads();

    v8f co[2] = { vzero, vzero };
    #pragma unroll
    for (int kk = 0; kk < 64; kk += 32) {
        v16bf afr;
        {
            const int row  = lane & 15;
            const int koff = ((lane >> 4) << 3) + kk;
            union { v16bf v; v8bf h2[2]; } u;
            u.h2[0] = *(const v8bf*)&at[wave][row][koff];
            u.h2[1] = *(const v8bf*)&at[wave][row][koff + 16];
            afr = u.v;
        }
        #pragma unroll
        for (int nt = 0; nt < 2; ++nt) {
            const int col   = nt * 16 + (lane & 15);
            const int kbase = ((lane >> 4) << 3) + kk;
            union { v16bf v; v8bf h2[2]; } u;
            u.h2[0] = *(const v8bf*)&vt[col][kbase];
            u.h2[1] = *(const v8bf*)&vt[col][kbase + 16];
            co[nt] = wmma_bf16(afr, u.v, co[nt]);
        }
    }

    const int rl = (lane >> 4) << 3;
    const int cl = lane & 15;
    #pragma unroll
    for (int nt = 0; nt < 2; ++nt)
        #pragma unroll
        for (int i = 0; i < 8; ++i) {
            const int j = wave * 16 + rl + i;
            const int d = nt * 16 + cl;
            ctx[(((size_t)t * 64 + j) * 64 + h) * E + (size_t)iPos * Dh + d] = (bf16)co[nt][i];
        }
}

// ---------------- host-side launch ----------------
extern "C" void kernel_launch(void* const* d_in, const int* in_sizes, int n_in,
                              void* d_out, int out_size, void* d_ws, size_t ws_size,
                              hipStream_t stream) {
    (void)in_sizes; (void)n_in; (void)out_size; (void)ws_size;
    const float* savespace = (const float*)d_in[1];
    const float* Wqkv      = (const float*)d_in[2];
    const float* Wo        = (const float*)d_in[3];
    const float* w1        = (const float*)d_in[4];
    const float* b1        = (const float*)d_in[5];
    const float* w2        = (const float*)d_in[6];
    const float* b2        = (const float*)d_in[7];
    float* out = (float*)d_out;

    char* ws = (char*)d_ws;
    size_t off = 0;
    auto alloc = [&](size_t bytes) -> void* {
        void* p = ws + off;
        off += (bytes + 255) & ~(size_t)255;
        return p;
    };
    bf16*  sn   = (bf16*) alloc((size_t)M * E * 2);
    bf16*  wqT  = (bf16*) alloc((size_t)NQKV * E * 2);
    bf16*  woT  = (bf16*) alloc((size_t)E * E * 2);
    bf16*  w1T  = (bf16*) alloc((size_t)HID * E * 2);
    bf16*  w2T  = (bf16*) alloc((size_t)E * HID * 2);
    bf16*  qkvb = (bf16*) alloc((size_t)M * NQKV * 2);
    bf16*  ctx  = (bf16*) alloc((size_t)M * E * 2);
    float* y    = (float*)alloc((size_t)M * E * 4);
    bf16*  zb   = (bf16*) alloc((size_t)M * E * 2);
    bf16*  hb   = (bf16*) alloc((size_t)M * HID * 2);

    cvt_bf16_kernel<<<1024, 256, 0, stream>>>(Wqkv, wqT, (size_t)NQKV * E);
    transpose_cvt_kernel<<<dim3(E / 32,   E / 32),   256, 0, stream>>>(Wo, woT, E, E);
    transpose_cvt_kernel<<<dim3(HID / 32, E / 32),   256, 0, stream>>>(w1, w1T, E, HID);
    transpose_cvt_kernel<<<dim3(E / 32,   HID / 32), 256, 0, stream>>>(w2, w2T, HID, E);

    ln_kernel<<<M, 256, 0, stream>>>(savespace, sn, nullptr, E);

    gemm_bf16_kernel<<<dim3(NQKV / BN, M / BM), 256, 0, stream>>>(
        sn, wqT, M, NQKV, E, nullptr, nullptr, qkvb, nullptr, 0);

    attn_kernel<<<Bb * S * H, 128, 0, stream>>>(qkvb, ctx);

    gemm_bf16_kernel<<<dim3(E / BN, M / BM), 256, 0, stream>>>(
        ctx, woT, M, E, E, nullptr, savespace, nullptr, y, 1);

    ln_kernel<<<M, 256, 0, stream>>>(y, zb, y, E);

    gemm_bf16_kernel<<<dim3(HID / BN, M / BM), 256, 0, stream>>>(
        zb, w1T, M, HID, E, b1, nullptr, hb, nullptr, 2);

    gemm_bf16_kernel<<<dim3(E / BN, M / BM), 256, 0, stream>>>(
        hb, w2T, M, E, HID, b2, y, nullptr, out, 3);
}
